// BatchFFTMA_11510512353806
// MI455X (gfx1250) — matrix operand order repaired and checked by reading, not compiled, not run on hardware
//
#include <hip/hip_runtime.h>
#include <hip/hip_bf16.h>

typedef __attribute__((ext_vector_type(16))) _Float16 v16h;
typedef __attribute__((ext_vector_type(8)))  float    v8f;

namespace {
constexpr int HH   = 96;
constexpr int WW   = 96;
constexpr int WIN  = 65;    // 2*D+1
constexpr int RAD  = 32;    // D
constexpr int NW2  = 160;   // padded noise row stride (H+2D)
constexpr int STR  = 98;    // LDS row stride in halfs (96 data + 2 pad, stride/2 odd)
constexpr int BROWS = 80;   // padded M/N extent (5 tiles of 16)
constexpr int BUF_H = BROWS * STR;
constexpr int NBUF  = 8;
constexpr float TWO_PI_65 = 6.283185307179586f / 65.0f;
}

__device__ __forceinline__ v8f wmma_f16(v16h a, v16h b, v8f c) {
  // D = A(16x32 f16) * B(32x16 f16) + C(f32)
  return __builtin_amdgcn_wmma_f32_16x16x32_f16(
      /*neg_a=*/false, a, /*neg_b=*/false, b,
      /*c_mod=*/(short)0, c, /*reuse_a=*/false, /*reuse_b=*/false);
}

// A fragment: lane holds row m = rowbase + (lane&15);
// K contiguous: lanes 0-15 -> kb..kb+15, lanes 16-31 -> kb+16..kb+31.
__device__ __forceinline__ v16h ldsA(const _Float16* m, int rowbase, int kbase, int lane) {
  const _Float16* p = m + (rowbase + (lane & 15)) * STR + kbase + ((lane >> 4) << 4);
  v16h a;
#pragma unroll
  for (int i = 0; i < 16; ++i) a[i] = p[i];
  return a;
}

// B fragment from column-major storage (mT[n][k]): lane holds col n = nbase+(lane&15),
// K contiguous: lanes 0-15 -> kb..kb+15, lanes 16-31 -> kb+16..kb+31.
__device__ __forceinline__ v16h ldsB(const _Float16* mT, int nbase, int kbase, int lane) {
  const _Float16* p = mT + (nbase + (lane & 15)) * STR + kbase + ((lane >> 4) << 4);
  v16h b;
#pragma unroll
  for (int i = 0; i < 16; ++i) b[i] = p[i];
  return b;
}

extern "C" __global__ __launch_bounds__(256)
void fftma_wmma_kernel(const float* __restrict__ angle,
                       const float* __restrict__ noise,
                       float* __restrict__ out) {
  extern __shared__ _Float16 smem[];
  _Float16* Cm  = smem + 0 * BUF_H;  // cos DFT matrix (symmetric)
  _Float16* Sm  = smem + 1 * BUF_H;  // sin DFT matrix (symmetric)
  _Float16* Xm  = smem + 2 * BUF_H;  // patch
  _Float16* Rm  = smem + 3 * BUF_H;  // covariance
  _Float16* U1T = smem + 4 * BUF_H;  // (X*C)^T
  _Float16* U2T = smem + 5 * BUF_H;  // (X*S)^T
  _Float16* T1T = smem + 6 * BUF_H;  // (R*C)^T
  _Float16* T2T = smem + 7 * BUF_H;  // (R*S)^T
  __shared__ float red[4];           // sumX, sumR, sumV2*N2, center*N2

  const int tid  = threadIdx.x;
  const int lane = tid & 31;
  const int w    = tid >> 5;
  const int p    = blockIdx.x;
  const int py   = p / WW;
  const int px   = p % WW;

  // ---- phase 0: zero LDS (padding regions must be exact zeros for GEMM) ----
  {
    uint32_t* z = (uint32_t*)smem;
    const int nz = (NBUF * BUF_H) / 2;
    for (int i = tid; i < nz; i += 256) z[i] = 0u;
    if (tid < 4) red[tid] = 0.f;
  }
  __syncthreads();

  // ---- phase 1: build C,S, patch X, covariance R; reduce sums ----
  float c0, s0;
  __sincosf(angle[p], &s0, &c0);
  float sX = 0.f, sR = 0.f;
  for (int idx = tid; idx < WIN * WIN; idx += 256) {
    const int i = idx / WIN, j = idx % WIN;
    // exact DFT angles via integer modulo (k*u mod 65)
    const int m65 = (i * j) % WIN;
    float sv, cv;
    __sincosf(TWO_PI_65 * (float)m65, &sv, &cv);
    Cm[i * STR + j] = (_Float16)cv;
    Sm[i * STR + j] = (_Float16)sv;
    // sliding-window patch
    const float xv = noise[(py + i) * NW2 + (px + j)];
    Xm[i * STR + j] = (_Float16)xv;
    sX += xv;
    // anisotropic covariance, EXP = 0.5 -> sqrt
    const float xu = (float)(i - RAD), yv = (float)(j - RAD);
    const float a1 = xu * c0 + yv * s0;
    const float b1 = -xu * s0 + yv * c0;
    const float q  = a1 * a1 * (1.f / 225.f) + b1 * b1 * (1.f / 9.f);
    const float rv = __expf(-sqrtf(q));
    Rm[i * STR + j] = (_Float16)rv;
    sR += rv;
  }
  atomicAdd(&red[0], sX);
  atomicAdd(&red[1], sR);
  __syncthreads();

  // ---- phase 2: U1=X*C, U2=X*S, T1=R*C, T2=R*S (stored transposed) ----
  // 4 products x 25 tiles = 100 wave-jobs.
  for (int job = w; job < 100; job += 8) {
    const int which = job / 25;
    const int t     = job % 25;
    const int tm    = (t / 5) * 16;
    const int tn    = (t % 5) * 16;
    const _Float16* Asrc = (which < 2) ? Xm : Rm;
    const _Float16* Bsrc = (which & 1) ? Sm : Cm;  // symmetric: row-major == col-major
    _Float16* Dst = (which == 0) ? U1T : (which == 1) ? U2T : (which == 2) ? T1T : T2T;
    v8f acc = {0.f, 0.f, 0.f, 0.f, 0.f, 0.f, 0.f, 0.f};
#pragma unroll
    for (int ks = 0; ks < 3; ++ks) {
      v16h a = ldsA(Asrc, tm, ks * 32, lane);
      v16h b = ldsB(Bsrc, tn, ks * 32, lane);
      acc = wmma_f16(a, b, acc);
    }
    const int n  = tn + (lane & 15);
    const int mb = tm + ((lane >> 4) << 3);
#pragma unroll
    for (int r = 0; r < 8; ++r)
      Dst[n * STR + (mb + r)] = (_Float16)acc[r];  // transposed store
  }
  __syncthreads();

  // ---- phase 3: Wf = (C-iS)X(C-iS)^T, Ft = FFT2(R); fused reductions ----
  float s2p = 0.f, cenp = 0.f;
  for (int t = w; t < 25; t += 8) {
    const int tm = (t / 5) * 16;
    const int tn = (t % 5) * 16;
    v8f pA = {0.f,0.f,0.f,0.f,0.f,0.f,0.f,0.f}, pB = pA, pC = pA, pD = pA;
    v8f qA = pA, qB = pA, qC = pA, qD = pA;
#pragma unroll
    for (int ks = 0; ks < 3; ++ks) {
      const int kb = ks * 32;
      v16h ac = ldsA(Cm, tm, kb, lane);
      v16h as = ldsA(Sm, tm, kb, lane);
      v16h b1 = ldsB(U1T, tn, kb, lane);
      v16h b2 = ldsB(U2T, tn, kb, lane);
      v16h b3 = ldsB(T1T, tn, kb, lane);
      v16h b4 = ldsB(T2T, tn, kb, lane);
      pA = wmma_f16(ac, b1, pA);  // C*U1
      pB = wmma_f16(as, b2, pB);  // S*U2
      pC = wmma_f16(ac, b2, pC);  // C*U2
      pD = wmma_f16(as, b1, pD);  // S*U1
      qA = wmma_f16(ac, b3, qA);  // C*T1
      qB = wmma_f16(as, b4, qB);  // S*T2
      qC = wmma_f16(ac, b4, qC);  // C*T2
      qD = wmma_f16(as, b3, qD);  // S*T1
    }
    const int n  = tn + (lane & 15);
    const int mb = tm + ((lane >> 4) << 3);
    if (n < WIN) {
#pragma unroll
      for (int r = 0; r < 8; ++r) {
        const int m = mb + r;
        if (m < WIN) {
          const float Wr = pA[r] - pB[r];          // Re Wf
          const float Wi = -(pC[r] + pD[r]);       // Im Wf
          const float Fr = qA[r] - qB[r];          // Re FFT2(R) (unshifted)
          const float Fi = -(qC[r] + qD[r]);       // Im FFT2(R)
          const int ph = (RAD * (m + n)) % WIN;    // phi = 2*pi*32*(k1+k2)/65
          float sph, cph;
          __sincosf(TWO_PI_65 * (float)ph, &sph, &cph);
          // fftshift phase rotation -> real Rf
          const float Rf = Fr * cph - Fi * sph;
          const float rp = Rf + 1e-8f;
          const float gp = fmaxf(rp, 0.f);
          const float gn = fmaxf(-rp, 0.f);
          s2p += (Wr * Wr + Wi * Wi) * gp;         // -> N2 * sum(v^2)
          const float Gr = sqrtf(gp), Gi = sqrtf(gn);   // complex sqrt parts
          const float ReF = Wr * Gr - Wi * Gi;
          const float ImF = Wr * Gi + Wi * Gr;
          cenp += ReF * cph - ImF * sph;           // -> N2 * v[D,D] (same phase)
        }
      }
    }
  }
#pragma unroll
  for (int off = 16; off > 0; off >>= 1) {
    s2p  += __shfl_xor(s2p, off);
    cenp += __shfl_xor(cenp, off);
  }
  if (lane == 0) {
    atomicAdd(&red[2], s2p);
    atomicAdd(&red[3], cenp);
  }
  __syncthreads();

  if (tid == 0) {
    const float N2   = (float)(WIN * WIN);          // 4225
    const float S1   = red[0] * sqrtf(red[1] + 1e-8f);  // sum(v) = Wf0 * G0
    const float mean = S1 / N2;
    const float S2   = red[2] / N2;                 // sum(v^2)
    const float vc   = red[3] / N2;                 // v[D,D]
    const float var  = (S2 - N2 * mean * mean) / (N2 - 1.f);  // ddof=1
    const float sd   = sqrtf(fmaxf(var, 0.f));
    out[p] = (vc - mean) / (sd + 1e-6f);
  }
}

extern "C" void kernel_launch(void* const* d_in, const int* in_sizes, int n_in,
                              void* d_out, int out_size, void* d_ws, size_t ws_size,
                              hipStream_t stream) {
  const float* angle = (const float*)d_in[0];  // [96,96] f32
  const float* noise = (const float*)d_in[1];  // [1,1,160,160] f32
  float* o = (float*)d_out;                    // [96,96] f32
  (void)in_sizes; (void)n_in; (void)out_size; (void)d_ws; (void)ws_size;
  const size_t lds_bytes = (size_t)NBUF * BUF_H * sizeof(_Float16);  // ~122.5 KB
  fftma_wmma_kernel<<<HH * WW, 256, lds_bytes, stream>>>(angle, noise, o);
}
